// GCN_42932493091126
// MI455X (gfx1250) — matrix operand order repaired and checked by reading, not compiled, not run on hardware
//
#include <hip/hip_runtime.h>
#include <hip/hip_bf16.h>

typedef __attribute__((ext_vector_type(16))) _Float16 v16h;
typedef __attribute__((ext_vector_type(8)))  float    v8f;
typedef unsigned int u32x4 __attribute__((ext_vector_type(4)));
typedef int          i32x8 __attribute__((ext_vector_type(8)));
typedef int          i32x4 __attribute__((ext_vector_type(4)));

constexpr int NUM_USER = 100000;
constexpr int NUM_ITEM = 50000;
constexpr int DIM      = 64;
constexpr int FEAT     = 4096;
constexpr int NN       = NUM_USER + NUM_ITEM;   // 150000
constexpr int HID      = 4 * DIM;               // 256

#if defined(__has_builtin)
#if __has_builtin(__builtin_amdgcn_tensor_load_to_lds) && __has_builtin(__builtin_amdgcn_s_wait_tensorcnt)
#define HAVE_TDM 1
#endif
#endif

// ---------------------------------------------------------------------------
// GEMM1: h_mlp = leaky_relu(features @ W1 + b1)   [50000 x 4096] @ [4096 x 256]
// block = 64 rows x 256 cols, 8 waves (2x4), wave tile = 32x64 (2x4 WMMA tiles)
// fp32 -> fp16 conversion fused into LDS staging; prefetch next K-chunk.
// ---------------------------------------------------------------------------
__global__ __launch_bounds__(256) void gemm1_kernel(
    const float* __restrict__ A, const float* __restrict__ W1,
    const float* __restrict__ b1, _Float16* __restrict__ out)
{
    __shared__ __align__(16) _Float16 sA[64 * 34];    // [row][k], padded stride
    __shared__ __align__(16) _Float16 sB[256 * 34];   // transposed [n][k]

    const int tid  = threadIdx.x;
    const int lane = tid & 31;
    const int wave = tid >> 5;
    const int wr   = wave >> 2;       // 0..1
    const int wc   = wave & 3;        // 0..3
    const int mBase = blockIdx.x * 64;

    const int lm  = lane & 15;
    const int kbA = (lane < 16) ? 0 : 8;    // A frag K-group base (16-bit A layout)
    const int kbB = (lane < 16) ? 0 : 8;    // B frag K-group base

    v8f zeroV;
#pragma unroll
    for (int e = 0; e < 8; ++e) zeroV[e] = 0.0f;
    v8f acc[2][4];
#pragma unroll
    for (int i = 0; i < 2; ++i)
#pragma unroll
        for (int j = 0; j < 4; ++j) acc[i][j] = zeroV;

    for (int kc = 0; kc < FEAT; kc += 32) {
        // --- stage A chunk: 64 rows x 32 k (fp32 -> fp16) ---
        {
            const int r  = tid >> 2;
            const int k0 = (tid & 3) * 8;
            const int m  = mBase + r;
            float4 f0, f1;
            if (m < NUM_ITEM) {
                const float4* p = reinterpret_cast<const float4*>(
                    A + (size_t)m * FEAT + kc + k0);
                f0 = p[0]; f1 = p[1];
                if (kc + 32 < FEAT)  // prefetch next K-chunk of A into caches
                    __builtin_prefetch(A + (size_t)m * FEAT + kc + 32 + k0, 0, 1);
            } else {
                f0 = make_float4(0.f, 0.f, 0.f, 0.f); f1 = f0;
            }
            _Float16* d = &sA[r * 34 + k0];
            d[0] = (_Float16)f0.x; d[1] = (_Float16)f0.y;
            d[2] = (_Float16)f0.z; d[3] = (_Float16)f0.w;
            d[4] = (_Float16)f1.x; d[5] = (_Float16)f1.y;
            d[6] = (_Float16)f1.z; d[7] = (_Float16)f1.w;
        }
        // --- stage B chunk transposed: sB[n][k], n = tid ---
#pragma unroll 8
        for (int kk = 0; kk < 32; ++kk)
            sB[tid * 34 + kk] = (_Float16)W1[(size_t)(kc + kk) * HID + tid];
        if (kc + 32 < FEAT) {   // prefetch next 32x256 W1 chunk (one line per thread)
            const int r2 = tid >> 3;          // 0..31
            const int c2 = (tid & 7) * 32;    // 0..224, 128B line granularity
            __builtin_prefetch(W1 + (size_t)(kc + 32 + r2) * HID + c2, 0, 1);
        }
        __syncthreads();

        // --- build A fragments (16x32 f16 per ISA layout) ---
        v16h af[2];
#pragma unroll
        for (int tr = 0; tr < 2; ++tr) {
            const _Float16* p = &sA[(wr * 32 + tr * 16 + lm) * 34 + kbA];
#pragma unroll
            for (int j = 0; j < 8; ++j) af[tr][j] = p[j];
#pragma unroll
            for (int j = 0; j < 8; ++j) af[tr][8 + j] = p[16 + j];
        }
        // --- 4 B fragments, 8 WMMAs ---
#pragma unroll
        for (int tc = 0; tc < 4; ++tc) {
            const _Float16* p = &sB[(wc * 64 + tc * 16 + lm) * 34 + kbB];
            v16h bf;
#pragma unroll
            for (int j = 0; j < 8; ++j) { bf[j] = p[j]; bf[8 + j] = p[16 + j]; }
            acc[0][tc] = __builtin_amdgcn_wmma_f32_16x16x32_f16(
                false, af[0], false, bf, (short)0, acc[0][tc], false, false);
            acc[1][tc] = __builtin_amdgcn_wmma_f32_16x16x32_f16(
                false, af[1], false, bf, (short)0, acc[1][tc], false, false);
        }
        __syncthreads();
    }

    // epilogue: bias + leaky_relu(0.01), store fp16
#pragma unroll
    for (int tr = 0; tr < 2; ++tr)
#pragma unroll
        for (int tc = 0; tc < 4; ++tc) {
            const int n = wc * 64 + tc * 16 + lm;
            const float bias = b1[n];
#pragma unroll
            for (int r = 0; r < 8; ++r) {
                const int m = mBase + wr * 32 + tr * 16 + ((lane < 16) ? r : r + 8);
                if (m < NUM_ITEM) {
                    float v = acc[tr][tc][r] + bias;
                    v = (v > 0.f) ? v : 0.01f * v;
                    out[(size_t)m * HID + n] = (_Float16)v;
                }
            }
        }
}

// ---------------------------------------------------------------------------
// GEMM2 compute step: one 128x32 fp16 A chunk (LDS) x resident W2 (LDS)
// ---------------------------------------------------------------------------
__device__ __forceinline__ void gemm2_compute(
    const _Float16* __restrict__ sAbuf, const _Float16* __restrict__ sB,
    v8f acc[4], int wave, int lm, int kbA, int kbB, int kc)
{
    v16h af;
    {
        const _Float16* p = &sAbuf[(wave * 16 + lm) * 34 + kbA];
#pragma unroll
        for (int j = 0; j < 8; ++j) af[j] = p[j];
#pragma unroll
        for (int j = 0; j < 8; ++j) af[8 + j] = p[16 + j];
    }
#pragma unroll
    for (int tc = 0; tc < 4; ++tc) {
        const _Float16* p = &sB[(tc * 16 + lm) * 258 + kc + kbB];
        v16h bf;
#pragma unroll
        for (int j = 0; j < 8; ++j) { bf[j] = p[j]; bf[8 + j] = p[16 + j]; }
        acc[tc] = __builtin_amdgcn_wmma_f32_16x16x32_f16(
            false, af, false, bf, (short)0, acc[tc], false, false);
    }
}

#if defined(HAVE_TDM)
// Issue a TDM load of a 128x32 fp16 tile (row stride HID elements) into LDS,
// with pad: +1 DWORD every 16 DWORDs -> LDS row stride 34 halves.
// tensor_dim1 = rows remaining => hardware zero-fills the M-tail.
__device__ __forceinline__ void gemm2_tdm_issue(
    const _Float16* gbase, unsigned lds_addr, int rows_remaining)
{
    const unsigned long long ga = (unsigned long long)(uintptr_t)gbase;
    const int tdim1 = rows_remaining < 128 ? rows_remaining : 128;
    u32x4 g0;
    g0[0] = 1u;                                            // count=1, user descriptor
    g0[1] = lds_addr;                                      // lds_addr [63:32]
    g0[2] = (unsigned)(ga & 0xffffffffull);                // global_addr lo
    g0[3] = (unsigned)((ga >> 32) & 0x1ffffffull) | (2u << 30); // addr hi | type=2
    i32x8 g1;
    g1[0] = (1 << 16) | (1 << 20) | (3 << 22);  // data_size=2B, pad_en, every 16DW +1DW
    g1[1] = (32 << 16);                         // tensor_dim0 = 32  (bits 79:48)
    g1[2] = (tdim1 << 16);                      // tensor_dim1 lo    (bits 111:80)
    g1[3] = (32 << 16);                         // tile_dim0 = 32    (bits 127:112)
    g1[4] = 128;                                // tile_dim1 = 128, tile_dim2 = 0
    g1[5] = HID;                                // tensor_dim0_stride = 256
    g1[6] = 0;
    g1[7] = 0;
    i32x4 z4; z4[0] = 0; z4[1] = 0; z4[2] = 0; z4[3] = 0;  // 2-D: groups 2/3 unused
    i32x8 z8;
#pragma unroll
    for (int j = 0; j < 8; ++j) z8[j] = 0;
    // 6-arg form (clang-23 / therock-10.0 headers)
    __builtin_amdgcn_tensor_load_to_lds(g0, g1, z4, z4, z8, 0);
}
#endif

// ---------------------------------------------------------------------------
// GEMM2: temp = h_mlp @ W2 + b2    [50000 x 256] @ [256 x 64]
// W2 fully LDS-resident (transposed). A chunks streamed by the Tensor Data
// Mover into a double buffer, overlapped with WMMA compute.
// ---------------------------------------------------------------------------
__global__ __launch_bounds__(256) void gemm2_kernel(
    const _Float16* __restrict__ A, const float* __restrict__ W2,
    const float* __restrict__ b2, float* __restrict__ xitems)
{
    __shared__ __align__(16) _Float16 sA[2][128 * 34];
    __shared__ __align__(16) _Float16 sB[64 * 258];   // [n][k], whole W2

    const int tid  = threadIdx.x;
    const int lane = tid & 31;
    const int wave = tid >> 5;
    const int mBase = blockIdx.x * 128;

    const int lm  = lane & 15;
    const int kbA = (lane < 16) ? 0 : 8;
    const int kbB = (lane < 16) ? 0 : 8;

    // load whole W2 transposed (256x64 fp32 -> [n][k] fp16)
#pragma unroll 4
    for (int i = 0; i < 64; ++i) {
        const int idx = i * 256 + tid;
        const int k = idx >> 6, n = idx & 63;
        sB[n * 258 + k] = (_Float16)W2[idx];
    }

    v8f acc[4];
#pragma unroll
    for (int j = 0; j < 4; ++j) {
#pragma unroll
        for (int e = 0; e < 8; ++e) acc[j][e] = 0.0f;
    }

#if defined(HAVE_TDM)
    const int rowsLeft = NUM_ITEM - mBase;
    const _Float16* gtile = A + (size_t)mBase * HID;
    if (wave == 0)   // prologue: chunk 0 -> buffer 0
        gemm2_tdm_issue(gtile, (unsigned)(uintptr_t)&sA[0][0], rowsLeft);

    for (int i = 0; i < HID / 32; ++i) {
        const int kc  = i * 32;
        const int buf = i & 1;
        __syncthreads();   // prior readers of sA[buf^1] are done
        if (wave == 0) {
            if (i + 1 < HID / 32) {
                gemm2_tdm_issue(gtile + kc + 32,
                                (unsigned)(uintptr_t)&sA[buf ^ 1][0], rowsLeft);
                __builtin_amdgcn_s_wait_tensorcnt(1);  // chunk i landed
            } else {
                __builtin_amdgcn_s_wait_tensorcnt(0);  // last chunk landed
            }
        }
        __syncthreads();   // publish chunk i to all waves
        gemm2_compute(&sA[buf][0], sB, acc, wave, lm, kbA, kbB, kc);
    }
#else
    // fallback: synchronous cooperative staging (single buffer)
    for (int kc = 0; kc < HID; kc += 32) {
        {
            const int r  = tid >> 1;
            const int k0 = (tid & 1) * 16;
            const int m  = mBase + r;
            uint4 u0, u1;
            if (m < NUM_ITEM) {
                const uint4* p = reinterpret_cast<const uint4*>(
                    A + (size_t)m * HID + kc + k0);
                u0 = p[0]; u1 = p[1];
            } else {
                u0 = make_uint4(0u, 0u, 0u, 0u); u1 = u0;
            }
            unsigned int* d = reinterpret_cast<unsigned int*>(&sA[0][r * 34 + k0]);
            d[0] = u0.x; d[1] = u0.y; d[2] = u0.z; d[3] = u0.w;
            d[4] = u1.x; d[5] = u1.y; d[6] = u1.z; d[7] = u1.w;
        }
        __syncthreads();
        gemm2_compute(&sA[0][0], sB, acc, wave, lm, kbA, kbB, kc);
        __syncthreads();
    }
#endif

#pragma unroll
    for (int tc = 0; tc < 4; ++tc) {
        const int n = tc * 16 + lm;
        const float bias = b2[n];
#pragma unroll
        for (int r = 0; r < 8; ++r) {
            const int m = mBase + wave * 16 + ((lane < 16) ? r : r + 8);
            if (m < NUM_ITEM)
                xitems[(size_t)m * DIM + n] = acc[tc][r] + bias;
        }
    }
}

// ---------------------------------------------------------------------------
// Row-wise L2 normalize: x = concat(preference, temp) / max(||row||, 1e-12)
// one 64-wide row per wave32 (2 floats per lane)
// ---------------------------------------------------------------------------
__global__ __launch_bounds__(256) void normalize_kernel(
    const float* __restrict__ pref, const float* __restrict__ xitems,
    float* __restrict__ xnorm)
{
    const int lane = threadIdx.x & 31;
    const int row  = blockIdx.x * 8 + (threadIdx.x >> 5);
    if (row >= NN) return;
    const float* src = (row < NUM_USER)
        ? (pref + (size_t)row * DIM)
        : (xitems + (size_t)(row - NUM_USER) * DIM);
    float2 v = reinterpret_cast<const float2*>(src)[lane];
    float s = v.x * v.x + v.y * v.y;
#pragma unroll
    for (int o = 16; o > 0; o >>= 1) s += __shfl_down(s, o, 32);
    s = __shfl(s, 0, 32);
    const float inv = 1.0f / fmaxf(sqrtf(s), 1e-12f);
    float2 o2; o2.x = v.x * inv; o2.y = v.y * inv;
    reinterpret_cast<float2*>(xnorm + (size_t)row * DIM)[lane] = o2;
}

__global__ void zero_kernel(float* __restrict__ p, size_t n)
{
    size_t i = (size_t)blockIdx.x * blockDim.x + threadIdx.x;
    const size_t stride = (size_t)gridDim.x * blockDim.x;
    for (; i < n; i += stride) p[i] = 0.0f;
}

__global__ void deg_kernel(const int* __restrict__ row, float* __restrict__ deg, int E)
{
    const int e = blockIdx.x * blockDim.x + threadIdx.x;
    if (e < E) atomicAdd(&deg[row[e]], 1.0f);
}

__global__ void dis_kernel(float* __restrict__ deg)
{
    const int i = blockIdx.x * blockDim.x + threadIdx.x;
    if (i < NN) {
        const float d = deg[i];
        deg[i] = (d > 0.f) ? (1.0f / sqrtf(d)) : 0.0f;   // isolated nodes never gathered
    }
}

// one edge per wave: dst[col] += dis[row]*dis[col] * src[row]
__global__ __launch_bounds__(256) void conv_kernel(
    const float* __restrict__ src, float* __restrict__ dst,
    const int* __restrict__ row, const int* __restrict__ col,
    const float* __restrict__ dis, int E)
{
    const int lane = threadIdx.x & 31;
    const int e = blockIdx.x * 8 + (threadIdx.x >> 5);
    if (e >= E) return;
    const int r = __builtin_amdgcn_readfirstlane(row[e]);   // wave-uniform -> SGPR
    const int c = __builtin_amdgcn_readfirstlane(col[e]);
    const float nrm = dis[r] * dis[c];
    float2 v = reinterpret_cast<const float2*>(src + (size_t)r * DIM)[lane];
    float* d = dst + (size_t)c * DIM + lane * 2;
    atomicAdd(d,     nrm * v.x);
    atomicAdd(d + 1, nrm * v.y);
}

// out[0:NX) = xnorm + h + h1 ; out[NX:) = preference
__global__ void final_kernel(
    const float* __restrict__ xnorm, const float* __restrict__ h,
    const float* __restrict__ h1, const float* __restrict__ pref,
    float* __restrict__ out, size_t total)
{
    const size_t NX = (size_t)NN * DIM;                // 9,600,000
    const size_t i = (size_t)blockIdx.x * blockDim.x + threadIdx.x;
    if (i >= total) return;
    if (i < NX) out[i] = xnorm[i] + h[i] + h1[i];
    else        out[i] = pref[i - NX];
}

// ---------------------------------------------------------------------------
extern "C" void kernel_launch(void* const* d_in, const int* in_sizes, int n_in,
                              void* d_out, int out_size, void* d_ws, size_t ws_size,
                              hipStream_t stream)
{
    const int*   edge = (const int*)  d_in[0];   // [2, E]
    const float* feat = (const float*)d_in[1];   // [50000, 4096]
    const float* pref = (const float*)d_in[2];   // [100000, 64]
    const float* W1   = (const float*)d_in[3];   // [4096, 256]
    const float* b1   = (const float*)d_in[4];   // [256]
    const float* W2   = (const float*)d_in[5];   // [256, 64]
    const float* b2   = (const float*)d_in[6];   // [64]
    float* out = (float*)d_out;

    const int E = in_sizes[0] / 2;               // 1,200,000
    const int* erow = edge;
    const int* ecol = edge + E;

    // workspace layout (bytes, 16B-aligned blocks)
    char* ws = (char*)d_ws;
    const size_t SZ_HMLP = (size_t)NUM_ITEM * HID * sizeof(_Float16); // 25,600,000
    const size_t SZ_DEG  = (size_t)NN * sizeof(float);                //    600,000
    const size_t SZ_X    = (size_t)NN * DIM * sizeof(float);          // 38,400,000
    _Float16* hmlp  = (_Float16*)(ws);
    float*    deg   = (float*)(ws + SZ_HMLP);                 // deg,h,h1 contiguous
    float*    h     = (float*)(ws + SZ_HMLP + SZ_DEG);
    float*    h1    = (float*)(ws + SZ_HMLP + SZ_DEG + SZ_X);
    float*    xnorm = (float*)(ws + SZ_HMLP + SZ_DEG + 2 * SZ_X);
    float*    xitem = (float*)(ws + SZ_HMLP + SZ_DEG + 3 * SZ_X);

    // 1) zero deg + h + h1 (contiguous)
    const size_t nzero = (SZ_DEG + 2 * SZ_X) / sizeof(float);
    zero_kernel<<<8192, 256, 0, stream>>>(deg, nzero);

    // 2) MLP
    gemm1_kernel<<<(NUM_ITEM + 63) / 64, 256, 0, stream>>>(feat, W1, b1, hmlp);
    gemm2_kernel<<<(NUM_ITEM + 127) / 128, 256, 0, stream>>>(hmlp, W2, b2, xitem);

    // 3) concat + row L2 normalize
    normalize_kernel<<<(NN + 7) / 8, 256, 0, stream>>>(pref, xitem, xnorm);

    // 4) degree -> deg^-1/2 (in place)
    deg_kernel<<<(E + 255) / 256, 256, 0, stream>>>(erow, deg, E);
    dis_kernel<<<(NN + 255) / 256, 256, 0, stream>>>(deg);

    // 5) two GCN propagation passes
    conv_kernel<<<(E + 7) / 8, 256, 0, stream>>>(xnorm, h,  erow, ecol, deg, E);
    conv_kernel<<<(E + 7) / 8, 256, 0, stream>>>(h,     h1, erow, ecol, deg, E);

    // 6) x_hat = x + h + h1 ; append preference
    const size_t total = (size_t)out_size;       // 16,000,000
    final_kernel<<<(unsigned)((total + 255) / 256), 256, 0, stream>>>(
        xnorm, h, h1, pref, out, total);
}
